// SLVM_75909251989878
// MI455X (gfx1250) — compile-verified, matrix operand order checked by reading, not run on hardware
//
#include <hip/hip_runtime.h>
#include <hip/hip_bf16.h>

typedef __attribute__((ext_vector_type(16))) _Float16 v16h;
typedef __attribute__((ext_vector_type(8)))  _Float16 v8h;
typedef __attribute__((ext_vector_type(8)))  float    v8f;

#define V16CAT(lo,hi) __builtin_shufflevector(lo,hi,0,1,2,3,4,5,6,7,8,9,10,11,12,13,14,15)

static constexpr int Bn   = 1024;
static constexpr int Tn   = 64;
static constexpr int FEAn = 256;
static constexpr int HIDn = 1024;
static constexpr int Z1n  = 64;
static constexpr int Z2n  = 128;
static constexpr int ACTn = 16;
static constexpr float EPSc   = 1e-6f;
static constexpr float SLOPEc = 0.2f;

// ---------------------------------------------------------------------------
// WMMA GEMM:  Y(M x N) = X(M x K) @ W  with W pre-transposed as Wt[N][K] (f16)
// M = gridDim.y * 128 (always 1024 here). Block = 256 threads = 8 waves.
// Wave grid 4(m) x 2(n); each wave computes a 32x32 tile = 2x2 WMMA frags.
// Double-buffered LDS: one barrier per 32-wide K step; tile i+1 staging
// (global->regs->LDS) overlaps tile i WMMAs.
// EPI==0: out = leaky_relu(acc + bias) -> f16   (hidden layers)
// EPI==1: out = acc + bias             -> f32   (final layer)
// ---------------------------------------------------------------------------
template<int EPI>
__global__ __launch_bounds__(256)
void gemm_wmma(const _Float16* __restrict__ X, int K,
               const _Float16* __restrict__ Wt,
               const float* __restrict__ bias,
               void* __restrict__ Yv, int N)
{
    constexpr int LDSTR = 40;   // row stride in halves: 80B, 16B-aligned, bank-skewed
    __shared__ alignas(16) _Float16 lsA[2][128 * LDSTR];
    __shared__ alignas(16) _Float16 lsB[2][64 * LDSTR];

    const int tid  = threadIdx.x;
    const int lane = tid & 31;
    const int wave = tid >> 5;
    const int wm   = wave >> 1;       // 0..3
    const int wn   = wave & 1;        // 0..1
    const int m_blk = blockIdx.y * 128;
    const int n_blk = blockIdx.x * 64;
    const int lm    = lane & 15;
    const int khalf = (lane >> 4) * 8;   // lanes 0-15 -> K 0..7/16..23 ; 16-31 -> 8..15/24..31

    // Per-thread staging addresses (A: 2 x 16B chunks, B: 1 x 16B chunk).
    const int aIdx0 = tid;             // chunks 0..255
    const int aIdx1 = tid + 256;       // chunks 256..511
    const int aRow0 = aIdx0 >> 2, aOff0 = (aIdx0 & 3) * 8;
    const int aRow1 = aIdx1 >> 2, aOff1 = (aIdx1 & 3) * 8;
    const int bRow  = tid >> 2,   bOff  = (tid & 3) * 8;

    const _Float16* gA0 = X  + (size_t)(m_blk + aRow0) * (size_t)K + aOff0;
    const _Float16* gA1 = X  + (size_t)(m_blk + aRow1) * (size_t)K + aOff1;
    const _Float16* gB  = Wt + (size_t)(n_blk + bRow ) * (size_t)K + bOff;

    // Prologue: stage K-tile 0 into buffer 0.
    v8h ra0 = *(const v8h*)gA0;
    v8h ra1 = *(const v8h*)gA1;
    v8h rb  = *(const v8h*)gB;
    *(v8h*)&lsA[0][aRow0 * LDSTR + aOff0] = ra0;
    *(v8h*)&lsA[0][aRow1 * LDSTR + aOff1] = ra1;
    *(v8h*)&lsB[0][bRow  * LDSTR + bOff ] = rb;
    __syncthreads();

    v8f c00 = {}, c01 = {}, c10 = {}, c11 = {};

    const int ar0 = (wm * 32 + lm) * LDSTR;
    const int ar1 = (wm * 32 + 16 + lm) * LDSTR;
    const int br0 = (wn * 32 + lm) * LDSTR;
    const int br1 = (wn * 32 + 16 + lm) * LDSTR;

    const int niter = K >> 5;
    for (int it = 0; it < niter; ++it) {
        const int cur  = it & 1;
        const int nxt  = cur ^ 1;
        const bool more = (it + 1) < niter;

        // Issue global loads for the next K-tile early (overlap with WMMAs).
        if (more) {
            const int k0 = (it + 1) << 5;
            ra0 = *(const v8h*)(gA0 + k0);
            ra1 = *(const v8h*)(gA1 + k0);
            rb  = *(const v8h*)(gB  + k0);
        }

        // Fragments from current buffer.
        v8h a0l = *(v8h*)&lsA[cur][ar0 + khalf], a0h = *(v8h*)&lsA[cur][ar0 + khalf + 16];
        v8h a1l = *(v8h*)&lsA[cur][ar1 + khalf], a1h = *(v8h*)&lsA[cur][ar1 + khalf + 16];
        v8h b0l = *(v8h*)&lsB[cur][br0 + khalf], b0h = *(v8h*)&lsB[cur][br0 + khalf + 16];
        v8h b1l = *(v8h*)&lsB[cur][br1 + khalf], b1h = *(v8h*)&lsB[cur][br1 + khalf + 16];

        v16h a0 = V16CAT(a0l, a0h);
        v16h a1 = V16CAT(a1l, a1h);
        v16h b0 = V16CAT(b0l, b0h);
        v16h b1 = V16CAT(b1l, b1h);

        c00 = __builtin_amdgcn_wmma_f32_16x16x32_f16(false, a0, false, b0, (short)0, c00, false, false);
        c01 = __builtin_amdgcn_wmma_f32_16x16x32_f16(false, a0, false, b1, (short)0, c01, false, false);
        c10 = __builtin_amdgcn_wmma_f32_16x16x32_f16(false, a1, false, b0, (short)0, c10, false, false);
        c11 = __builtin_amdgcn_wmma_f32_16x16x32_f16(false, a1, false, b1, (short)0, c11, false, false);

        // Stage next tile into the other buffer (no conflict with current reads).
        if (more) {
            *(v8h*)&lsA[nxt][aRow0 * LDSTR + aOff0] = ra0;
            *(v8h*)&lsA[nxt][aRow1 * LDSTR + aOff1] = ra1;
            *(v8h*)&lsB[nxt][bRow  * LDSTR + bOff ] = rb;
        }
        __syncthreads();   // single barrier per K-step
    }

    // Epilogue. C/D layout: VGPR r -> M=r (lanes 0-15) or M=8+r (lanes 16-31); N=lane&15.
    const int gm = m_blk + wm * 32 + ((lane >> 4) * 8);
    const int gn = n_blk + wn * 32 + lm;
    const float bv0 = bias[gn];
    const float bv1 = bias[gn + 16];

    auto emit = [&](int m, int n, float v) {
        if (EPI == 0) {
            v = (v > 0.f) ? v : SLOPEc * v;
            ((_Float16*)Yv)[(size_t)m * (size_t)N + n] = (_Float16)v;
        } else {
            ((float*)Yv)[(size_t)m * (size_t)N + n] = v;
        }
    };

    #pragma unroll
    for (int r = 0; r < 8; ++r) {
        emit(gm + r,      gn,      c00[r] + bv0);
        emit(gm + r,      gn + 16, c01[r] + bv1);
        emit(gm + 16 + r, gn,      c10[r] + bv0);
        emit(gm + 16 + r, gn + 16, c11[r] + bv1);
    }
}

// ---------------------------------------------------------------------------
// Weight convert + transpose: W[K][N] (f32, row-major) -> Wt[N][Kpad] (f16),
// zero-padded in K so the WMMA K-loop (step 32) needs no edge handling.
// ---------------------------------------------------------------------------
__global__ void wconv(const float* __restrict__ W, _Float16* __restrict__ Wt,
                      int K, int N, int Kpad)
{
    int i = blockIdx.x * 256 + threadIdx.x;
    if (i >= N * Kpad) return;
    int n = i / Kpad;
    int k = i - n * Kpad;
    Wt[i] = (k < K) ? (_Float16)W[(size_t)k * N + n] : (_Float16)0.f;
}

// ---------------------------------------------------------------------------
// Build concatenated f16 input X[B][Kpad] from up to 4 f32 sources (row strides
// per source), zero-padding the tail.
// ---------------------------------------------------------------------------
__global__ void build_x(_Float16* __restrict__ X, int Kpad,
                        const float* s0, int w0, int st0,
                        const float* s1, int w1, int st1,
                        const float* s2, int w2, int st2,
                        const float* s3, int w3, int st3)
{
    int i = blockIdx.x * 256 + threadIdx.x;
    if (i >= Bn * Kpad) return;
    int b = i / Kpad;
    int k = i - b * Kpad;
    float v = 0.f;
    if (k < w0)                     v = s0[(size_t)b * st0 + k];
    else if (k < w0 + w1)           v = s1[(size_t)b * st1 + (k - w0)];
    else if (k < w0 + w1 + w2)      v = s2[(size_t)b * st2 + (k - w0 - w1)];
    else if (k < w0 + w1 + w2 + w3) v = s3[(size_t)b * st3 + (k - w0 - w1 - w2)];
    X[i] = (_Float16)v;
}

static __device__ __forceinline__ float softplusf(float x) {
    return fmaxf(x, 0.f) + log1pf(expf(-fabsf(x)));
}

// t = 0: KL(q || N(0,1)), z1 = mu_q ; initializes klacc.
__global__ void kl0_z1(const float* __restrict__ OUT, float* __restrict__ z1,
                       float* __restrict__ klacc, float* __restrict__ aux_out)
{
    int b = blockIdx.x * 256 + threadIdx.x;
    if (b >= Bn) return;
    float acc = 0.f;
    for (int d = 0; d < Z1n; ++d) {
        float mq = OUT[(size_t)b * 128 + d];
        float sq = softplusf(OUT[(size_t)b * 128 + 64 + d]) + EPSc;
        acc += -logf(sq) + 0.5f * (sq * sq + mq * mq) - 0.5f;
        z1[(size_t)b * Z1n + d] = mq;
        aux_out[(size_t)b * Tn * 192 + d] = mq;
    }
    klacc[b] = acc * (1.f / Z1n);
}

// t >= 1: KL(q || p), z1 = mu_q ; accumulates into klacc.
__global__ void kl_step(const float* __restrict__ OUTQ, const float* __restrict__ OUTP,
                        float* __restrict__ z1, float* __restrict__ klacc,
                        float* __restrict__ aux_out, int t)
{
    int b = blockIdx.x * 256 + threadIdx.x;
    if (b >= Bn) return;
    float acc = 0.f;
    for (int d = 0; d < Z1n; ++d) {
        float mq = OUTQ[(size_t)b * 128 + d];
        float sq = softplusf(OUTQ[(size_t)b * 128 + 64 + d]) + EPSc;
        float mp = OUTP[(size_t)b * 128 + d];
        float sp = softplusf(OUTP[(size_t)b * 128 + 64 + d]) + EPSc;
        float dm = mq - mp;
        acc += (logf(sp) - logf(sq)) + (sq * sq + dm * dm) / (2.f * sp * sp) - 0.5f;
        z1[(size_t)b * Z1n + d] = mq;
        aux_out[(size_t)b * Tn * 192 + (size_t)t * 192 + d] = mq;
    }
    klacc[b] += acc * (1.f / Z1n);
}

// z2 = mu part of gen_z2 output (first 128 of 256); also scatter to aux_z_seq.
__global__ void z2_write(const float* __restrict__ OUTG, float* __restrict__ z2,
                         float* __restrict__ aux_out, int t)
{
    int i = blockIdx.x * 256 + threadIdx.x;
    if (i >= Bn * Z2n) return;
    int b = i / Z2n;
    int d = i - b * Z2n;
    float v = OUTG[(size_t)b * 256 + d];
    z2[i] = v;
    aux_out[(size_t)b * Tn * 192 + (size_t)t * 192 + 64 + d] = v;
}

__global__ void kl_final(const float* __restrict__ klacc, float* __restrict__ KLout)
{
    int b = blockIdx.x * 256 + threadIdx.x;
    if (b >= Bn) return;
    KLout[b] = klacc[b] * (1.f / Tn);
}

// ---------------------------------------------------------------------------
// Host orchestration
// ---------------------------------------------------------------------------
extern "C" void kernel_launch(void* const* d_in, const int* in_sizes, int n_in,
                              void* d_out, int out_size, void* d_ws, size_t ws_size,
                              hipStream_t stream)
{
    (void)in_sizes; (void)n_in; (void)out_size; (void)ws_size;

    const float* aux = (const float*)d_in[0];   // (B, T, FEA)
    const float* act = (const float*)d_in[1];   // (B, T-1, ACT)

    // 5 MLPs x (W1,b1,W2,b2,W3,b3)
    const float* P[5][6];
    {
        int idx = 2;
        for (int m = 0; m < 5; ++m)
            for (int j = 0; j < 6; ++j)
                P[m][j] = (const float*)d_in[idx++];
    }
    static const int DIN[5]  = {256, 320, 80, 336, 464};
    static const int KPAD[5] = {256, 320, 96, 352, 480};
    static const int DOUT[5] = {128, 256, 128, 128, 256};

    // Workspace carve-out
    char*  ws  = (char*)d_ws;
    size_t off = 0;
    auto alloc = [&](size_t bytes) -> void* {
        void* p = ws + off;
        off = (off + bytes + 255) & ~(size_t)255;
        return p;
    };

    _Float16* Wt[5][3];
    for (int m = 0; m < 5; ++m) {
        Wt[m][0] = (_Float16*)alloc((size_t)HIDn * KPAD[m] * 2);
        Wt[m][1] = (_Float16*)alloc((size_t)HIDn * HIDn * 2);
        Wt[m][2] = (_Float16*)alloc((size_t)DOUT[m] * HIDn * 2);
    }
    _Float16* X    = (_Float16*)alloc((size_t)Bn * 480 * 2);
    _Float16* H1   = (_Float16*)alloc((size_t)Bn * HIDn * 2);
    _Float16* H2   = (_Float16*)alloc((size_t)Bn * HIDn * 2);
    float*    OUTP = (float*)alloc((size_t)Bn * 128 * 4);
    float*    OUTQ = (float*)alloc((size_t)Bn * 128 * 4);
    float*    OUTG = (float*)alloc((size_t)Bn * 256 * 4);
    float*    z1   = (float*)alloc((size_t)Bn * Z1n * 4);
    float*    z2   = (float*)alloc((size_t)Bn * Z2n * 4);
    float*    klac = (float*)alloc((size_t)Bn * 4);

    float* KLout   = (float*)d_out;
    float* aux_out = (float*)d_out + Bn;   // (B, T, 192)

    auto cdiv = [](int a, int b) { return (a + b - 1) / b; };

    // 1) Convert weights once per launch (deterministic; reused across 64 steps)
    for (int m = 0; m < 5; ++m) {
        wconv<<<cdiv(HIDn * KPAD[m], 256), 256, 0, stream>>>(P[m][0], Wt[m][0], DIN[m], HIDn, KPAD[m]);
        wconv<<<cdiv(HIDn * HIDn, 256), 256, 0, stream>>>(P[m][2], Wt[m][1], HIDn, HIDn, HIDn);
        wconv<<<cdiv(DOUT[m] * HIDn, 256), 256, 0, stream>>>(P[m][4], Wt[m][2], HIDn, DOUT[m], HIDn);
    }

    // 3-layer MLP: X(Kpad) -> H1 -> H2 -> OUT(f32, width dout)
    auto mlp = [&](int m, int Kp, float* OUT) {
        gemm_wmma<0><<<dim3(HIDn / 64, Bn / 128), 256, 0, stream>>>(X, Kp, Wt[m][0], P[m][1], H1, HIDn);
        gemm_wmma<0><<<dim3(HIDn / 64, Bn / 128), 256, 0, stream>>>(H1, HIDn, Wt[m][1], P[m][3], H2, HIDn);
        gemm_wmma<1><<<dim3(DOUT[m] / 64, Bn / 128), 256, 0, stream>>>(H2, HIDn, Wt[m][2], P[m][5], OUT, DOUT[m]);
    };

    const int AUXS = Tn * FEAn;         // aux row stride
    const int ACTS = (Tn - 1) * ACTn;   // action row stride

    // ---- t = 0 ----
    build_x<<<cdiv(Bn * 256, 256), 256, 0, stream>>>(X, 256,
        aux, 256, AUXS, nullptr, 0, 0, nullptr, 0, 0, nullptr, 0, 0);
    mlp(0, 256, OUTQ);
    kl0_z1<<<cdiv(Bn, 256), 256, 0, stream>>>(OUTQ, z1, klac, aux_out);

    build_x<<<cdiv(Bn * 320, 256), 256, 0, stream>>>(X, 320,
        aux, 256, AUXS, z1, 64, 64, nullptr, 0, 0, nullptr, 0, 0);
    mlp(1, 320, OUTG);
    z2_write<<<cdiv(Bn * Z2n, 256), 256, 0, stream>>>(OUTG, z2, aux_out, 0);

    // ---- t = 1 .. 63 ----
    for (int t = 1; t < Tn; ++t) {
        const float* feat_t = aux + (size_t)t * FEAn;
        const float* act_t  = act + (size_t)(t - 1) * ACTn;

        // prior: concat(z1, act) -> gen_z1_t
        build_x<<<cdiv(Bn * 96, 256), 256, 0, stream>>>(X, 96,
            z1, 64, 64, act_t, 16, ACTS, nullptr, 0, 0, nullptr, 0, 0);
        mlp(2, 96, OUTP);

        // posterior: concat(z1, act, feat_t) -> inf_z1_t
        build_x<<<cdiv(Bn * 352, 256), 256, 0, stream>>>(X, 352,
            z1, 64, 64, act_t, 16, ACTS, feat_t, 256, AUXS, nullptr, 0, 0);
        mlp(3, 352, OUTQ);

        kl_step<<<cdiv(Bn, 256), 256, 0, stream>>>(OUTQ, OUTP, z1, klac, aux_out, t);

        // gen z2: concat(z2_old, act, feat_t, z1_new) -> gen_z2_t
        build_x<<<cdiv(Bn * 480, 256), 256, 0, stream>>>(X, 480,
            z2, 128, 128, act_t, 16, ACTS, feat_t, 256, AUXS, z1, 64, 64);
        mlp(4, 480, OUTG);
        z2_write<<<cdiv(Bn * Z2n, 256), 256, 0, stream>>>(OUTG, z2, aux_out, t);
    }

    kl_final<<<cdiv(Bn, 256), 256, 0, stream>>>(klac, KLout);
}
